// WeaveNetHead_46875273068814
// MI455X (gfx1250) — matrix operand order repaired and checked by reading, not compiled, not run on hardware
//
#include <hip/hip_runtime.h>

// ---------------- types ----------------
typedef __attribute__((ext_vector_type(16))) __bf16 v16bf;
typedef __attribute__((ext_vector_type(8)))  __bf16 bf16x8;
typedef __attribute__((ext_vector_type(8)))  float  v8f;

#define KP 136   // padded K stride (bf16 elems) in LDS, 272B rows (16B aligned)

__device__ __forceinline__ __bf16 to_bf16(float f) {
    unsigned u = __builtin_bit_cast(unsigned, f);
    unsigned r = u + 0x7FFFu + ((u >> 16) & 1u);   // round-to-nearest-even
    unsigned short h = (unsigned short)(r >> 16);
    return __builtin_bit_cast(__bf16, h);
}

// monotonic float<->uint keys so atomicMax(uint) == float max
__device__ __forceinline__ unsigned fkey(float f) {
    unsigned u = __builtin_bit_cast(unsigned, f);
    return (u & 0x80000000u) ? ~u : (u | 0x80000000u);
}
__device__ __forceinline__ float funkey(unsigned k) {
    unsigned u = (k & 0x80000000u) ? (k ^ 0x80000000u) : ~k;
    return __builtin_bit_cast(float, u);
}

// Build a 16xbf16 fragment from two contiguous 8-element LDS chunks.
__device__ __forceinline__ v16bf frag16(const __bf16* p0, const __bf16* p1) {
    bf16x8 a = *reinterpret_cast<const bf16x8*>(p0);
    bf16x8 b = *reinterpret_cast<const bf16x8*>(p1);
    v16bf r;
#pragma unroll
    for (int i = 0; i < 8; ++i) { r[i] = a[i]; r[i + 8] = b[i]; }
    return r;
}

// Async-DMA one 16B chunk from global (bf16 data) into LDS; ASYNCcnt-tracked.
__device__ __forceinline__ void async_b128(unsigned lds_addr, const __bf16* g) {
    unsigned long long ga = (unsigned long long)g;
    asm volatile("global_load_async_to_lds_b128 %0, %1, off"
                 :: "v"(lds_addr), "v"(ga) : "memory");
}
__device__ __forceinline__ void wait_async0() {
    asm volatile("s_wait_asynccnt 0x0" ::: "memory");
}

// Stage 128 rows x Kc (bf16) of z = concat(zf, zs) into zsh via async DMA.
// Two threads per row: thread parity selects the source half.
__device__ __forceinline__ void stage_z(__bf16* zsh,
                                        const __bf16* zf, const __bf16* zs,
                                        int t, int b, int fixed, int chx,
                                        int halfC, int pool_inner) {
    int r = t >> 1, src = t & 1;
    int row = chx * 128 + r;
    long re = pool_inner ? (((long)b * 256 + fixed) * 256 + row)
                         : (((long)b * 256 + row) * 256 + fixed);
    const __bf16* p = (src ? zs : zf) + re * halfC;
    // generic LDS pointer: low 32 bits are the LDS byte address (aperture rule)
    unsigned lds = (unsigned)(unsigned long long)&zsh[r * KP + src * halfC];
    for (int k = 0; k < halfC; k += 8)                 // 16B per transfer
        async_b128(lds + (unsigned)(k * 2), p + k);
    wait_async0();
}

// ---------------------------------------------------------------------------
// GEMM1 (h = z@W1 + b1) + max-pool over the set axis + q = p @ W2_bottom.
// One workgroup (256 thr = 8 wave32) per (b, fixed). 256 set rows in two
// 128-row chunks. pool_inner=1: rows are m (pool over M, stream ab);
// pool_inner=0: rows are n (pool over N, stream ba_t).
// ---------------------------------------------------------------------------
__global__ __launch_bounds__(256)
void k_enc_pool(const __bf16* __restrict__ zf, const __bf16* __restrict__ zs,
                const float* __restrict__ W1, const float* __restrict__ bia,
                const float* __restrict__ W2, float* __restrict__ qout,
                int Kc, int pool_inner)
{
    __shared__ __bf16 zsh[128 * KP];
    __shared__ __bf16 wsh[64 * KP];
    __shared__ unsigned pool[64];
    __shared__ float pcol[64];

    const int t = threadIdx.x;
    const int b = blockIdx.y, fixed = blockIdx.x;
    const int halfC = Kc >> 1;

    if (t < 64) {
        pool[t] = 0u;
        __builtin_prefetch(W2 + (long)Kc * 64 + t, 0, 1);   // q-phase weights
        // stage W1^T in bf16: wsh[c*KP + k] = W1[k,c]
        for (int k = 0; k < Kc; ++k)
            wsh[t * KP + k] = to_bf16(W1[(long)k * 64 + t]);
    }

    const int wv = t >> 5, ln = t & 31, lq = ln & 15, lh = ln >> 4;
    float bc[4];
#pragma unroll
    for (int ct = 0; ct < 4; ++ct) bc[ct] = bia[ct * 16 + lq];

    for (int chx = 0; chx < 2; ++chx) {
        __syncthreads();
        stage_z(zsh, zf, zs, t, b, fixed, chx, halfC, pool_inner);
        __syncthreads();

        // each wave computes one 16-row tile x 4 channel tiles
        v8f acc[4];
#pragma unroll
        for (int ct = 0; ct < 4; ++ct)
#pragma unroll
            for (int i = 0; i < 8; ++i) acc[ct][i] = bc[ct];

        for (int kk = 0; kk < Kc; kk += 32) {
            // A fragment: row = lq, K chunks at kk+lh*8 and kk+16+lh*8
            const __bf16* ap = &zsh[(wv * 16 + lq) * KP + kk + lh * 8];
            v16bf A = frag16(ap, ap + 16);
#pragma unroll
            for (int ct = 0; ct < 4; ++ct) {
                // B fragment: col = lq, K chunks at kk+lh*8 and kk+16+lh*8
                const __bf16* bp = &wsh[(ct * 16 + lq) * KP + kk + lh * 8];
                v16bf B = frag16(bp, bp + 16);
                acc[ct] = __builtin_amdgcn_wmma_f32_16x16x32_bf16(
                    false, A, false, B, (short)0, acc[ct], false, false);
            }
        }

        // pool: lane covers 8 rows of channel c = ct*16+lq
#pragma unroll
        for (int ct = 0; ct < 4; ++ct) {
            float m = acc[ct][0];
#pragma unroll
            for (int i = 1; i < 8; ++i) m = fmaxf(m, acc[ct][i]);
            atomicMax(&pool[ct * 16 + lq], fkey(m));
        }
    }
    __syncthreads();
    if (t < 64) pcol[t] = funkey(pool[t]);
    __syncthreads();
    if (t < 64) { // q[j] = sum_k p[k] * W2[Kc+k, j]
        float s = 0.f;
        for (int k = 0; k < 64; ++k)
            s += pcol[k] * W2[(long)(Kc + k) * 64 + t];
        qout[((long)b * 256 + fixed) * 64 + t] = s;
    }
}

// ---------------------------------------------------------------------------
// GEMM2: y = z @ W2_top + q (broadcast along the pooled axis), plus per-channel
// BN partial sums (sum, sumsq) accumulated into global atomics.
// ---------------------------------------------------------------------------
__global__ __launch_bounds__(256)
void k_gemm2(const __bf16* __restrict__ zf, const __bf16* __restrict__ zs,
             const float* __restrict__ W2, const float* __restrict__ qin,
             float* __restrict__ dst, float* __restrict__ gsum,
             float* __restrict__ gsq, int Kc, int pool_inner)
{
    __shared__ __bf16 zsh[128 * KP];
    __shared__ __bf16 wsh[64 * KP];
    __shared__ float qrow[64];
    __shared__ float ssum[64], ssq[64];

    const int t = threadIdx.x;
    const int b = blockIdx.y, fixed = blockIdx.x;
    const int halfC = Kc >> 1;

    if (t < 64) {
        ssum[t] = 0.f; ssq[t] = 0.f;
        qrow[t] = qin[((long)b * 256 + fixed) * 64 + t];
        for (int k = 0; k < Kc; ++k)   // stage W2_top^T in bf16
            wsh[t * KP + k] = to_bf16(W2[(long)k * 64 + t]);
    }

    const int wv = t >> 5, ln = t & 31, lq = ln & 15, lh = ln >> 4;

    for (int chx = 0; chx < 2; ++chx) {
        __syncthreads();
        stage_z(zsh, zf, zs, t, b, fixed, chx, halfC, pool_inner);
        __syncthreads();

        v8f acc[4];
#pragma unroll
        for (int ct = 0; ct < 4; ++ct) {
            float q = qrow[ct * 16 + lq];
#pragma unroll
            for (int i = 0; i < 8; ++i) acc[ct][i] = q;
        }

        for (int kk = 0; kk < Kc; kk += 32) {
            const __bf16* ap = &zsh[(wv * 16 + lq) * KP + kk + lh * 8];
            v16bf A = frag16(ap, ap + 16);
#pragma unroll
            for (int ct = 0; ct < 4; ++ct) {
                const __bf16* bp = &wsh[(ct * 16 + lq) * KP + kk + lh * 8];
                v16bf B = frag16(bp, bp + 16);
                acc[ct] = __builtin_amdgcn_wmma_f32_16x16x32_bf16(
                    false, A, false, B, (short)0, acc[ct], false, false);
            }
        }

        // store y + BN partial sums (lane: channel c, rows mbase..mbase+7)
        int mbase = chx * 128 + wv * 16 + lh * 8;
#pragma unroll
        for (int ct = 0; ct < 4; ++ct) {
            int c = ct * 16 + lq;
            float s = 0.f, s2 = 0.f;
#pragma unroll
            for (int r = 0; r < 8; ++r) {
                float v = acc[ct][r];
                long row = mbase + r;
                long e = pool_inner
                    ? ((((long)b * 256 + fixed) * 256 + row) * 64 + c)
                    : ((((long)b * 256 + row) * 256 + fixed) * 64 + c);
                dst[e] = v;
                s += v; s2 += v * v;
            }
            atomicAdd(&ssum[c], s);
            atomicAdd(&ssq[c], s2);
        }
    }
    __syncthreads();
    if (t < 64) {
        atomicAdd(&gsum[t], ssum[t]);
        atomicAdd(&gsq[t],  ssq[t]);
    }
}

// stats[0..127]=sum (2 streams x 64ch), [128..255]=sumsq -> scale/shift
__global__ void k_finalize(const float* __restrict__ gsum, const float* __restrict__ gsq,
                           const float* __restrict__ gamma, const float* __restrict__ beta,
                           float* __restrict__ gscale, float* __restrict__ gshift,
                           float invN)
{
    int i = threadIdx.x;           // 0..127
    int c = i & 63;
    float m = gsum[i] * invN;
    float v = gsq[i] * invN - m * m;
    float sc = gamma[c] * rsqrtf(v + 1e-5f);
    gscale[i] = sc;
    gshift[i] = beta[c] - m * sc;
}

// in-place BN affine + PReLU (f32), plus bf16 mirror for next layer's GEMMs
__global__ __launch_bounds__(256)
void k_bnprelu(float* __restrict__ y, __bf16* __restrict__ mir,
               const float* __restrict__ sc, const float* __restrict__ sh,
               const float* __restrict__ ap)
{
    long i = ((long)blockIdx.x * 256 + threadIdx.x) * 4;
    float a = ap[0];
    float4 v = *reinterpret_cast<float4*>(y + i);
    int c = (int)(i & 63);
    float r0 = v.x * sc[c]     + sh[c];
    float r1 = v.y * sc[c + 1] + sh[c + 1];
    float r2 = v.z * sc[c + 2] + sh[c + 2];
    float r3 = v.w * sc[c + 3] + sh[c + 3];
    v.x = r0 >= 0.f ? r0 : a * r0;
    v.y = r1 >= 0.f ? r1 : a * r1;
    v.z = r2 >= 0.f ? r2 : a * r2;
    v.w = r3 >= 0.f ? r3 : a * r3;
    *reinterpret_cast<float4*>(y + i) = v;
    mir[i]     = to_bf16(v.x); mir[i + 1] = to_bf16(v.y);
    mir[i + 2] = to_bf16(v.z); mir[i + 3] = to_bf16(v.w);
}

// one-time f32 -> bf16 conversion for layer-0 inputs
__global__ __launch_bounds__(256)
void k_cvt(const float* __restrict__ x, __bf16* __restrict__ y)
{
    long i = ((long)blockIdx.x * 256 + threadIdx.x) * 4;
    float4 v = *reinterpret_cast<const float4*>(x + i);
    y[i]     = to_bf16(v.x); y[i + 1] = to_bf16(v.y);
    y[i + 2] = to_bf16(v.z); y[i + 3] = to_bf16(v.w);
}

// ---------------------------------------------------------------------------
extern "C" void kernel_launch(void* const* d_in, const int* in_sizes, int n_in,
                              void* d_out, int out_size, void* d_ws, size_t ws_size,
                              hipStream_t stream)
{
    const float* xab = (const float*)d_in[0];
    const float* xba = (const float*)d_in[1];
    const float* W1f = (const float*)d_in[2];
    const float* W1r = (const float*)d_in[3];
    const float* b1  = (const float*)d_in[4];
    const float* W2f = (const float*)d_in[5];
    const float* W2r = (const float*)d_in[6];
    const float* gam = (const float*)d_in[7];
    const float* bet = (const float*)d_in[8];
    const float* pa  = (const float*)d_in[9];

    const long S = 2L * 256 * 256 * 64;   // elems per stream (8,388,608)
    float* outAB = (float*)d_out;
    float* outBA = (float*)d_out + S;

    float* ws    = (float*)d_ws;
    float* bufAB = ws;
    float* bufBA = ws + S;
    float* qAB   = ws + 2 * S;
    float* qBA   = qAB + 2L * 256 * 64;
    float* stats = qBA + 2L * 256 * 64;
    float* gsum   = stats;         // [2][64]
    float* gsq    = stats + 128;   // [2][64]
    float* gscale = stats + 256;   // [2][64]
    float* gshift = stats + 384;   // [2][64]
    __bf16* mirAB = (__bf16*)(stats + 512);   // bf16 activation mirrors
    __bf16* mirBA = mirAB + S;

    const float invN = 1.0f / (float)(2 * 256 * 256);

    // layer-0 inputs -> bf16 mirrors (32 channels each)
    const long S0 = 2L * 256 * 256 * 32;
    k_cvt<<<(unsigned)(S0 / 1024), 256, 0, stream>>>(xab, mirAB);
    k_cvt<<<(unsigned)(S0 / 1024), 256, 0, stream>>>(xba, mirBA);

    dim3 grid(256, 2), blk(256);
    for (int l = 0; l < 6; ++l) {
        int Kc = l ? 128 : 64;
        const float* W1  = l ? (W1r + (long)(l - 1) * 128 * 64) : W1f;
        const float* W2  = l ? (W2r + (long)(l - 1) * 192 * 64) : W2f;
        const float* b1l = b1 + l * 64;
        float* dstAB = (l & 1) ? outAB : bufAB;
        float* dstBA = (l & 1) ? outBA : bufBA;

        hipMemsetAsync(gsum, 0, 256 * sizeof(float), stream);

        // ab: z=[xab,xba], pool over M (inner); ba: z=[xba,xab], pool over N
        k_enc_pool<<<grid, blk, 0, stream>>>(mirAB, mirBA, W1, b1l, W2, qAB, Kc, 1);
        k_enc_pool<<<grid, blk, 0, stream>>>(mirBA, mirAB, W1, b1l, W2, qBA, Kc, 0);
        k_gemm2<<<grid, blk, 0, stream>>>(mirAB, mirBA, W2, qAB, dstAB, gsum,      gsq,      Kc, 1);
        k_gemm2<<<grid, blk, 0, stream>>>(mirBA, mirAB, W2, qBA, dstBA, gsum + 64, gsq + 64, Kc, 0);
        k_finalize<<<1, 128, 0, stream>>>(gsum, gsq, gam + l * 64, bet + l * 64,
                                          gscale, gshift, invN);
        // normalize in place + refresh bf16 mirrors for the next layer
        k_bnprelu<<<(unsigned)(S / 1024), blk, 0, stream>>>(dstAB, mirAB, gscale,      gshift,      pa + l);
        k_bnprelu<<<(unsigned)(S / 1024), blk, 0, stream>>>(dstBA, mirBA, gscale + 64, gshift + 64, pa + l);
    }
}
